// ExtremaAwareSparseLayer_30562987278743
// MI455X (gfx1250) — matrix operand order repaired and checked by reading, not compiled, not run on hardware
//
#include <hip/hip_runtime.h>

// y = x @ (W*mask)^T + b  — M=16384, N=2048, K=2048, f32 in/out.
// Split-bf16 (hi+lo Dekker truncation split) GEMM on v_wmma_f32_16x16x32_bf16.
// MODE 2: x and W*mask pre-split into packed bf16 hi/lo in d_ws -> hot loop is
//         pure b128 copy staging + ds_load_b128 fragments + WMMA.
// MODE 1: only W*mask pre-split (16 MB ws).
// MODE 0: no ws; mask-mul + split done during staging.
// Addressing: uniform (scalar) base advanced per k-step + thread-constant
// 32-bit voffsets so staging loads use saddr+voffset form (no per-iter u64 VALU).

#define IN_F   2048
#define OUT_F  2048
#define BATCH  16384

#define BM 128
#define BN 128
#define BK 32
#define NK (IN_F / BK)
#define KPACK (IN_F / 2)              // packed u32 (2 bf16) per row

#define ROW_U32  20                   // 16 packed u32 + pad: 16B aligned rows, banks tiled
#define TILE_U32 (128 * ROW_U32)
#define BUF_U32  (4 * TILE_U32)       // Xh, Xl, Wh, Wl

typedef __attribute__((ext_vector_type(16))) __bf16   v16bf;
typedef __attribute__((ext_vector_type(8)))  float    v8f;
typedef __attribute__((ext_vector_type(4)))  unsigned v4u;

struct U8pair { v4u lo, hi; };        // 32B bit_cast target for v16bf

__device__ __forceinline__ unsigned pack_bf16(float a, float b) {
    const unsigned ua = __builtin_bit_cast(unsigned, a);
    const unsigned ub = __builtin_bit_cast(unsigned, b);
    return (ub & 0xFFFF0000u) | (ua >> 16);     // {lo16: trunc(a), hi16: trunc(b)}
}
__device__ __forceinline__ float hi_f32(float a) {
    return __builtin_bit_cast(float, __builtin_bit_cast(unsigned, a) & 0xFFFF0000u);
}
__device__ __forceinline__ v16bf load_frag(const unsigned* p, int second_off) {
    U8pair t;
    t.lo = *(const v4u*)(p);
    t.hi = *(const v4u*)(p + second_off);
    return __builtin_bit_cast(v16bf, t);
}
__device__ __forceinline__ void split4(const float4 v, uint2& h, uint2& l) {
    h.x = pack_bf16(v.x, v.y);
    h.y = pack_bf16(v.z, v.w);
    l.x = pack_bf16(v.x - hi_f32(v.x), v.y - hi_f32(v.y));   // exact residual, then trunc
    l.y = pack_bf16(v.z - hi_f32(v.z), v.w - hi_f32(v.w));
}

// ---------------- pre-pass kernels ----------------
__global__ __launch_bounds__(256)
void pack_w_kernel(const float* __restrict__ w, const float* __restrict__ mask,
                   unsigned* __restrict__ wh, unsigned* __restrict__ wl) {
    const size_t q = (size_t)blockIdx.x * 256 + threadIdx.x;   // one float4
    const float4 wv = *(const float4*)(w + q * 4);
    const float4 mv = *(const float4*)(mask + q * 4);
    float4 a;
    a.x = wv.x * mv.x; a.y = wv.y * mv.y; a.z = wv.z * mv.z; a.w = wv.w * mv.w;
    uint2 h, l;
    split4(a, h, l);
    *(uint2*)(wh + q * 2) = h;
    *(uint2*)(wl + q * 2) = l;
}

__global__ __launch_bounds__(256)
void pack_x_kernel(const float* __restrict__ x,
                   unsigned* __restrict__ xh, unsigned* __restrict__ xl) {
    const size_t q = (size_t)blockIdx.x * 256 + threadIdx.x;   // one float4
    const float4 v = *(const float4*)(x + q * 4);
    uint2 h, l;
    split4(v, h, l);
    *(uint2*)(xh + q * 2) = h;
    *(uint2*)(xl + q * 2) = l;
}

// ---------------- GEMM ----------------
template <int MODE>
__global__ __launch_bounds__(256)
void sparse_linear_wmma(const float* __restrict__ x,
                        const unsigned* __restrict__ xh, const unsigned* __restrict__ xl,
                        const float* __restrict__ w, const float* __restrict__ mask,
                        const unsigned* __restrict__ wh, const unsigned* __restrict__ wl,
                        const float* __restrict__ bias, float* __restrict__ y) {
    __shared__ unsigned sm[2 * BUF_U32];                 // 80 KB double buffer

    const int tid      = threadIdx.x;
    const int lane16   = tid & 15;
    const int laneHalf = (tid >> 4) & 1;
    const int wid      = tid >> 5;        // 0..7
    const int wm       = wid >> 2;        // 0..1 -> 64 M-rows / wave
    const int wn       = wid & 3;         // 0..3 -> 32 N-cols / wave
    const int mBase    = blockIdx.y * BM;
    const int nBase    = blockIdx.x * BN;

    v8f acc[4][2];
#pragma unroll
    for (int mt = 0; mt < 4; ++mt)
#pragma unroll
        for (int nt = 0; nt < 2; ++nt)
#pragma unroll
            for (int e = 0; e < 8; ++e) acc[mt][nt][e] = 0.0f;

    // ---- thread-constant 32-bit offsets (computed once, live in VGPRs) ----
    unsigned go2[2], lo2[2];              // packed-plane staging (r = idx>>2)
    unsigned goF[4], loF[4];              // f32-plane staging    (r = idx>>3)
#pragma unroll
    for (int i = 0; i < 2; ++i) {
        const int idx = tid + i * 256;
        const int r = idx >> 2, c2 = (idx & 3) << 2;
        go2[i] = (unsigned)(r * KPACK + c2);
        lo2[i] = (unsigned)(r * ROW_U32 + c2);
    }
#pragma unroll
    for (int i = 0; i < 4; ++i) {
        const int idx = tid + i * 256;
        const int r = idx >> 3, c = (idx & 7) << 2;
        goF[i] = (unsigned)(r * IN_F + c);
        loF[i] = (unsigned)(r * ROW_U32 + (c >> 1));
    }

    // staging registers (unused sets optimized away per MODE)
    float4 xf[4];
    v4u    xhv[2], xlv[2];
    float4 wf[4], mf[4];
    v4u    whv[2], wlv[2];

    auto stage_load = [&](int k0) {
        if constexpr (MODE == 2) {
            const unsigned* xhb = xh + (size_t)mBase * KPACK + (k0 >> 1);  // uniform
            const unsigned* xlb = xl + (size_t)mBase * KPACK + (k0 >> 1);
#pragma unroll
            for (int i = 0; i < 2; ++i) {
                xhv[i] = *(const v4u*)(xhb + go2[i]);
                xlv[i] = *(const v4u*)(xlb + go2[i]);
            }
        } else {
            const float* xb = x + (size_t)mBase * IN_F + k0;               // uniform
#pragma unroll
            for (int i = 0; i < 4; ++i) xf[i] = *(const float4*)(xb + goF[i]);
        }
        if constexpr (MODE >= 1) {
            const unsigned* whb = wh + (size_t)nBase * KPACK + (k0 >> 1);  // uniform
            const unsigned* wlb = wl + (size_t)nBase * KPACK + (k0 >> 1);
#pragma unroll
            for (int i = 0; i < 2; ++i) {
                whv[i] = *(const v4u*)(whb + go2[i]);
                wlv[i] = *(const v4u*)(wlb + go2[i]);
            }
        } else {
            const float* wb = w + (size_t)nBase * IN_F + k0;               // uniform
            const float* mb = mask + (size_t)nBase * IN_F + k0;
#pragma unroll
            for (int i = 0; i < 4; ++i) {
                wf[i] = *(const float4*)(wb + goF[i]);
                mf[i] = *(const float4*)(mb + goF[i]);
            }
        }
    };

    auto stage_store = [&](unsigned* buf) {
        unsigned* Xh = buf;
        unsigned* Xl = buf + TILE_U32;
        unsigned* Wh = buf + 2 * TILE_U32;
        unsigned* Wl = buf + 3 * TILE_U32;
        if constexpr (MODE == 2) {
#pragma unroll
            for (int i = 0; i < 2; ++i) {
                *(v4u*)(Xh + lo2[i]) = xhv[i];
                *(v4u*)(Xl + lo2[i]) = xlv[i];
            }
        } else {
#pragma unroll
            for (int i = 0; i < 4; ++i) {
                uint2 h, l;
                split4(xf[i], h, l);
                *(uint2*)(Xh + loF[i]) = h;
                *(uint2*)(Xl + loF[i]) = l;
            }
        }
        if constexpr (MODE >= 1) {
#pragma unroll
            for (int i = 0; i < 2; ++i) {
                *(v4u*)(Wh + lo2[i]) = whv[i];
                *(v4u*)(Wl + lo2[i]) = wlv[i];
            }
        } else {
#pragma unroll
            for (int i = 0; i < 4; ++i) {
                float4 a;
                a.x = wf[i].x * mf[i].x; a.y = wf[i].y * mf[i].y;
                a.z = wf[i].z * mf[i].z; a.w = wf[i].w * mf[i].w;
                uint2 h, l;
                split4(a, h, l);
                *(uint2*)(Wh + loF[i]) = h;
                *(uint2*)(Wl + loF[i]) = l;
            }
        }
    };

    auto compute = [&](const unsigned* buf) {
        const unsigned* Xh = buf;
        const unsigned* Xl = buf + TILE_U32;
        const unsigned* Wh = buf + 2 * TILE_U32;
        const unsigned* Wl = buf + 3 * TILE_U32;
        v16bf ah[4], al[4], bh[2], bl[2];
#pragma unroll
        for (int mt = 0; mt < 4; ++mt) {
            // A 16x32 bf16: lanes<16 K pairs {0,2,4,6},{16,18,20,22}; lanes>=16 +8
            const int off = (wm * 64 + mt * 16 + lane16) * ROW_U32 + 4 * laneHalf;
            ah[mt] = load_frag(Xh + off, 8);
            al[mt] = load_frag(Xl + off, 8);
        }
#pragma unroll
        for (int nt = 0; nt < 2; ++nt) {
            // B 32x16 bf16: N=lane16; lanes<16 K=0..7,16..23, lanes>=16 K=8..15,24..31
            const int off = (wn * 32 + nt * 16 + lane16) * ROW_U32 + 4 * laneHalf;
            bh[nt] = load_frag(Wh + off, 8);
            bl[nt] = load_frag(Wl + off, 8);
        }
#pragma unroll
        for (int mt = 0; mt < 4; ++mt)
#pragma unroll
            for (int nt = 0; nt < 2; ++nt) {
                acc[mt][nt] = __builtin_amdgcn_wmma_f32_16x16x32_bf16(
                    false, ah[mt], false, bh[nt], (short)0, acc[mt][nt], false, false);
                acc[mt][nt] = __builtin_amdgcn_wmma_f32_16x16x32_bf16(
                    false, ah[mt], false, bl[nt], (short)0, acc[mt][nt], false, false);
                acc[mt][nt] = __builtin_amdgcn_wmma_f32_16x16x32_bf16(
                    false, al[mt], false, bh[nt], (short)0, acc[mt][nt], false, false);
            }
    };

    // prologue: fill buffer 0
    stage_load(0);
    stage_store(sm);

    for (int it = 0; it < NK; ++it) {
        __syncthreads();
        const int p = it & 1;
        const bool more = (it + 1) < NK;
        if (more) stage_load((it + 1) * BK);     // HBM loads issued before compute
        compute(sm + p * BUF_U32);               // 24 WMMAs hide the latency
        if (more) stage_store(sm + (1 - p) * BUF_U32);
    }

    // epilogue: bias + store per C/D 16x16 f32 layout (VGPR v: M=v+8*laneHalf, N=lane16)
#pragma unroll
    for (int nt = 0; nt < 2; ++nt) {
        const int col  = nBase + wn * 32 + nt * 16 + lane16;
        const float bv = bias[col];
#pragma unroll
        for (int mt = 0; mt < 4; ++mt) {
            const int row0 = mBase + wm * 64 + mt * 16 + 8 * laneHalf;
#pragma unroll
            for (int v = 0; v < 8; ++v) {
                y[(size_t)(row0 + v) * OUT_F + col] = acc[mt][nt][v] + bv;
            }
        }
    }
}

extern "C" void kernel_launch(void* const* d_in, const int* in_sizes, int n_in,
                              void* d_out, int out_size, void* d_ws, size_t ws_size,
                              hipStream_t stream) {
    (void)in_sizes; (void)n_in; (void)out_size;
    const float* x    = (const float*)d_in[0];
    const float* w    = (const float*)d_in[1];
    const float* bias = (const float*)d_in[2];
    const float* mask = (const float*)d_in[3];
    float* y = (float*)d_out;

    const size_t wElems = (size_t)OUT_F * IN_F / 2;      // packed u32 per W plane
    const size_t xElems = (size_t)BATCH * IN_F / 2;      // packed u32 per X plane
    const size_t wBytes = 2 * wElems * sizeof(unsigned); // 16 MB
    const size_t xBytes = 2 * xElems * sizeof(unsigned); // 128 MB

    const dim3 grid(OUT_F / BN, BATCH / BM);             // (16, 128)
    const dim3 block(256);

    if (ws_size >= wBytes + xBytes) {
        unsigned* wh  = (unsigned*)d_ws;
        unsigned* wl  = wh + wElems;
        unsigned* xhp = wl + wElems;
        unsigned* xlp = xhp + xElems;
        pack_w_kernel<<<(OUT_F * IN_F / 4) / 256, block, 0, stream>>>(w, mask, wh, wl);
        pack_x_kernel<<<(BATCH * IN_F / 4) / 256, block, 0, stream>>>(x, xhp, xlp);
        sparse_linear_wmma<2><<<grid, block, 0, stream>>>(x, xhp, xlp, w, mask, wh, wl, bias, y);
    } else if (ws_size >= wBytes) {
        unsigned* wh = (unsigned*)d_ws;
        unsigned* wl = wh + wElems;
        pack_w_kernel<<<(OUT_F * IN_F / 4) / 256, block, 0, stream>>>(w, mask, wh, wl);
        sparse_linear_wmma<1><<<grid, block, 0, stream>>>(x, nullptr, nullptr, w, mask, wh, wl, bias, y);
    } else {
        sparse_linear_wmma<0><<<grid, block, 0, stream>>>(x, nullptr, nullptr, w, mask, nullptr, nullptr, bias, y);
    }
}
